// TransformerFlashback_72292889527144
// MI455X (gfx1250) — matrix operand order repaired and checked by reading, not compiled, not run on hardware
//
#include <hip/hip_runtime.h>
#include <hip/hip_bf16.h>
#include <math.h>

typedef __attribute__((ext_vector_type(16))) _Float16 v16h;
typedef __attribute__((ext_vector_type(8)))  _Float16 v8h;
typedef __attribute__((ext_vector_type(8)))  float    v8f;
typedef __attribute__((ext_vector_type(4)))  int      v4i;

static constexpr int S  = 128;
static constexpr int U  = 64;
static constexpr int D  = 512;
static constexpr int H  = 8;
static constexpr int HD = 64;
static constexpr int L  = 2;
static constexpr int V  = 20000;

// ---------------------------------------------------------------------------
// CDNA5 async copy: global -> LDS, 16 bytes per lane, tracked with ASYNCcnt.
// Builtin signature (from hipcc diagnostic): (v4i AS1*, v4i AS3*, Ii, Ii).
// ---------------------------------------------------------------------------
__device__ __forceinline__ void async_cp16(const _Float16* g, _Float16* l) {
#if __has_builtin(__builtin_amdgcn_global_load_async_to_lds_b128)
  __builtin_amdgcn_global_load_async_to_lds_b128(
      (__attribute__((address_space(1))) v4i*)(v4i*)(void*)const_cast<_Float16*>(g),
      (__attribute__((address_space(3))) v4i*)(v4i*)(void*)l,
      0, 0);
#else
  __attribute__((address_space(3))) _Float16* lp =
      (__attribute__((address_space(3))) _Float16*)l;
  unsigned long long ga = (unsigned long long)g;
  asm volatile("global_load_async_to_lds_b128 %0, %1, off"
               :: "v"(lp), "v"(ga) : "memory");
#endif
}

__device__ __forceinline__ void wait_async3() {
#if __has_builtin(__builtin_amdgcn_s_wait_asynccnt)
  __builtin_amdgcn_s_wait_asynccnt(3);
#else
  asm volatile("s_wait_asynccnt 0x3" ::: "memory");
#endif
}
__device__ __forceinline__ void wait_async0() {
#if __has_builtin(__builtin_amdgcn_s_wait_asynccnt)
  __builtin_amdgcn_s_wait_asynccnt(0);
#else
  asm volatile("s_wait_asynccnt 0x0" ::: "memory");
#endif
}

// ---------------------------------------------------------------------------
// Generic batched f16 WMMA GEMM:  C[b] = alpha * A[b] (MxK) @ B[b]^T (KxN) + bias
// A row-major MxK (f16), B row-major NxK (f16).  Per-batch offsets computed as
// (b/div)*so + (b%div)*si  (element units) for A, B, C.
// 256-thread block = 8 waves = 2x4 grid of 16x16 tiles (32x64 per block).
// K-step = 64; A(32x64) + B(64x64) tiles double-buffered in LDS, filled with
// global_load_async_to_lds_b128 (3 per thread per step, uniform ASYNCcnt).
// LDS row stride 72 halves (144B) to spread ds_load_b128 banks.
// Requires: K % 64 == 0, M % 32 == 0 covered by grid, N % 16 == 0.
// ---------------------------------------------------------------------------
static constexpr int LSTR = 72;  // halves per LDS tile row

__global__ void __launch_bounds__(256) wmma_gemm_f16(
    const _Float16* __restrict__ A, int adiv, long aso, long asi, int lda,
    const _Float16* __restrict__ B, int bdiv, long bso, long bsi, int ldb,
    float* __restrict__ C, _Float16* __restrict__ C16,
    int cdiv, long cso, long csi, int ldc,
    const float* __restrict__ bias,
    int M, int N, int K, float alpha, int relu)
{
  __shared__ _Float16 Abuf[2][32 * LSTR];
  __shared__ _Float16 Bbuf[2][64 * LSTR];

  int b = blockIdx.z;
  A += (long)(b / adiv) * aso + (long)(b % adiv) * asi;
  B += (long)(b / bdiv) * bso + (long)(b % bdiv) * bsi;
  long coff = (long)(b / cdiv) * cso + (long)(b % cdiv) * csi;

  int tid  = threadIdx.x;
  int wave = tid >> 5;
  int lane = tid & 31;
  int lh   = lane >> 4;                     // 0: lanes 0-15, 1: lanes 16-31
  int m0blk = blockIdx.y * 32;
  int n0blk = blockIdx.x * 64;
  int m0w = (wave >> 2) * 16;               // wave tile inside block
  int n0w = (wave & 3) * 16;
  bool valid = (m0blk + m0w + 16 <= M) && (n0blk + n0w + 16 <= N);

  // async-copy work split: A = 32 rows x 8 chunks (1/thread),
  //                        B = 64 rows x 8 chunks (2/thread, contiguous pair)
  int arow = tid >> 3;          // 0..31
  int akc  = (tid & 7) * 8;     // half-element offset within row
  int agr  = m0blk + arow; if (agr > M - 1) agr = M - 1;
  int brow = tid >> 2;          // 0..63
  int bkc  = (tid & 3) * 16;    // two 8-half chunks: bkc, bkc+8
  int bgr  = n0blk + brow; if (bgr > N - 1) bgr = N - 1;
  const _Float16* Agbase = A + (long)agr * lda + akc;
  const _Float16* Bgbase = B + (long)bgr * ldb + bkc;

  auto issue = [&](int kt, int buf) {
    async_cp16(Agbase + kt * 64, &Abuf[buf][arow * LSTR + akc]);
    async_cp16(Bgbase + kt * 64, &Bbuf[buf][brow * LSTR + bkc]);
    async_cp16(Bgbase + kt * 64 + 8, &Bbuf[buf][brow * LSTR + bkc + 8]);
  };

  const _Float16* Ar0 = &Abuf[0][(m0w + (lane & 15)) * LSTR];
  const _Float16* Br0 = &Bbuf[0][(n0w + (lane & 15)) * LSTR];
  const _Float16* Ar1 = &Abuf[1][(m0w + (lane & 15)) * LSTR];
  const _Float16* Br1 = &Bbuf[1][(n0w + (lane & 15)) * LSTR];

  v8f acc = {};
  int nsteps = K >> 6;
  issue(0, 0);
  for (int t = 0; t < nsteps; ++t) {
    int cb = t & 1;
    if (t + 1 < nsteps) {
      issue(t + 1, cb ^ 1);
      wait_async3();            // step-t copies (older) guaranteed complete
    } else {
      wait_async0();
    }
    __syncthreads();            // all waves' tile data visible
    const _Float16* Ar = cb ? Ar1 : Ar0;
    const _Float16* Br = cb ? Br1 : Br0;
#pragma unroll
    for (int ks = 0; ks < 64; ks += 32) {
      // A 16x32 fragment (ISA 7.12.2): lanes 0-15 K[ks..+7]&K[ks+16..+23],
      // lanes 16-31 K[ks+8..+15]&K[ks+24..+31]
      v8h a0 = *(const v8h*)(Ar + ks + lh * 8);
      v8h a1 = *(const v8h*)(Ar + ks + 16 + lh * 8);
      // B 32x16 fragment: same K split as A
      v8h b0 = *(const v8h*)(Br + ks + lh * 8);
      v8h b1 = *(const v8h*)(Br + ks + 16 + lh * 8);
      v16h av, bv;
#pragma unroll
      for (int i = 0; i < 8; ++i) {
        av[i] = a0[i]; av[i + 8] = a1[i];
        bv[i] = b0[i]; bv[i + 8] = b1[i];
      }
      acc = __builtin_amdgcn_wmma_f32_16x16x32_f16(
          false, av, false, bv, (short)0, acc, false, false);
    }
    __syncthreads();            // done reading buf cb before it is refilled
  }

  if (valid) {
    int cn = n0blk + n0w + (lane & 15);
    float bb = bias ? bias[cn] : 0.0f;
#pragma unroll
    for (int r = 0; r < 8; ++r) {
      int cm = m0blk + m0w + r + lh * 8;    // C/D: VGPR r -> M=r (+8 hi lanes)
      float v = acc[r] * alpha + bb;
      if (relu) v = v > 0.0f ? v : 0.0f;
      long idx = coff + (long)cm * ldc + cn;
      if (C)   C[idx]   = v;
      if (C16) C16[idx] = (_Float16)v;
    }
  }
}

// ---------------------------------------------------------------------------
__global__ void f32_to_f16_k(const float* __restrict__ in, _Float16* __restrict__ out, long n) {
  long i = (long)blockIdx.x * blockDim.x + threadIdx.x;
  long stride = (long)gridDim.x * blockDim.x;
  for (; i < n; i += stride) out[i] = (_Float16)in[i];
}

// user_pref[u] = sum_e inter_vals[au*50+e] * loc_emb[inter_cols[au*50+e]]
__global__ void user_pref_k(const int* __restrict__ active_user,
                            const int* __restrict__ inter_cols,
                            const float* __restrict__ inter_vals,
                            const float* __restrict__ loc_emb,
                            float* __restrict__ user_pref) {
  int u = blockIdx.x;
  int au = active_user[u];
  const int*   cols = inter_cols + (long)au * 50;
  const float* vals = inter_vals + (long)au * 50;
  int d = threadIdx.x;
  float acc = 0.0f;
  for (int e = 0; e < 50; ++e)
    acc += vals[e] * loc_emb[(long)cols[e] * D + d];
  user_pref[(long)u * D + d] = acc;
}

// Per (s,u) row: x_emb via 16-edge gather, src = x_emb + time_emb + s@coord_w + b,
// sim[u,s] = exp(-||user_pref[u]-x_emb||)
__global__ void src_k(const int* __restrict__ x, const int* __restrict__ t_slot,
                      const float* __restrict__ scoord,
                      const int* __restrict__ graph_cols, const float* __restrict__ graph_vals,
                      const float* __restrict__ loc_emb, const float* __restrict__ time_emb,
                      const float* __restrict__ coord_w, const float* __restrict__ coord_b,
                      const float* __restrict__ user_pref,
                      float* __restrict__ h32, _Float16* __restrict__ h16,
                      float* __restrict__ sim) {
  __shared__ float sm[D];
  int r = blockIdx.x;            // r = s*U + u
  int u = r % U, si = r / U;
  int d = threadIdx.x;
  int xi = x[r];
  const int*   cols = graph_cols + (long)xi * 16;
  const float* vals = graph_vals + (long)xi * 16;
  float xe = 0.0f;
  for (int e = 0; e < 16; ++e)
    xe += vals[e] * loc_emb[(long)cols[e] * D + d];

  float diff = user_pref[(long)u * D + d] - xe;
  sm[d] = diff * diff;
  __syncthreads();
  for (int off = D / 2; off > 0; off >>= 1) {
    if (d < off) sm[d] += sm[d + off];
    __syncthreads();
  }
  if (d == 0) sim[(long)u * S + si] = expf(-sqrtf(sm[0]));

  float srcv = xe + time_emb[(long)t_slot[r] * D + d]
             + scoord[(long)r * 2 + 0] * coord_w[d]
             + scoord[(long)r * 2 + 1] * coord_w[D + d]
             + coord_b[d];
  h32[(long)r * D + d] = srcv;
  h16[(long)r * D + d] = (_Float16)srcv;
}

// bias[u,i,j] = j>i ? -inf : alpha*ft(dt)*fs(ds)*sim[u,j]
__global__ void bias_k(const float* __restrict__ t, const float* __restrict__ scoord,
                       const float* __restrict__ sim, const float* __restrict__ alpha,
                       float* __restrict__ bias) {
  int i = blockIdx.x, u = blockIdx.y, j = threadIdx.x;
  float b;
  if (j > i) {
    b = -INFINITY;
  } else {
    float dt = t[(long)i * U + u] - t[(long)j * U + u];
    float dx = scoord[((long)i * U + u) * 2 + 0] - scoord[((long)j * U + u) * 2 + 0];
    float dy = scoord[((long)i * U + u) * 2 + 1] - scoord[((long)j * U + u) * 2 + 1];
    float ds = sqrtf(dx * dx + dy * dy);
    float ft = (cosf(dt * (float)(2.0 * M_PI / 86400.0)) + 1.0f) * 0.5f
             * expf(-dt * (0.1f / 86400.0f));
    float fs = expf(-ds * 100.0f);
    b = alpha[0] * ft * fs * sim[(long)u * S + j];
  }
  bias[((long)u * S + i) * S + j] = b;
}

// repack V slice of qkv16 (S,U,3D) -> vT (U,H,HD,S)
__global__ void repackV_k(const _Float16* __restrict__ qkv16, _Float16* __restrict__ vT) {
  int r = blockIdx.x;            // r = s*U + u
  int u = r % U, si = r / U;
  int e = threadIdx.x;           // 0..511
  int h = e >> 6, d = e & 63;
  vT[(((long)(u * H + h)) * HD + d) * S + si] =
      qkv16[(long)r * (3 * D) + 2 * D + e];
}

// softmax over j of scores*scale + bias  -> attn (f16)
__global__ void softmax_k(const float* __restrict__ scores, const float* __restrict__ bias,
                          _Float16* __restrict__ attn, float scale) {
  __shared__ float sm[S];
  int i = blockIdx.x, b = blockIdx.y;      // b = u*H + h
  int u = b >> 3;
  int j = threadIdx.x;
  long off = ((long)b * S + i) * S;
  float v = scores[off + j] * scale + bias[((long)u * S + i) * S + j];
  sm[j] = v; __syncthreads();
  for (int o = S / 2; o > 0; o >>= 1) {
    if (j < o) sm[j] = fmaxf(sm[j], sm[j + o]);
    __syncthreads();
  }
  float mx = sm[0]; __syncthreads();
  float e = expf(v - mx);
  sm[j] = e; __syncthreads();
  for (int o = S / 2; o > 0; o >>= 1) {
    if (j < o) sm[j] += sm[j + o];
    __syncthreads();
  }
  attn[off + j] = (_Float16)(e / sm[0]);
}

// h = LN(hin + delta)*g + b ; writes f32 + f16
__global__ void ln_k(const float* __restrict__ hin, const float* __restrict__ delta,
                     const float* __restrict__ g, const float* __restrict__ bb,
                     float* __restrict__ hout, _Float16* __restrict__ h16) {
  __shared__ float sm[D];
  int r = blockIdx.x, d = threadIdx.x;
  float xv = hin[(long)r * D + d] + delta[(long)r * D + d];
  sm[d] = xv; __syncthreads();
  for (int o = D / 2; o > 0; o >>= 1) {
    if (d < o) sm[d] += sm[d + o];
    __syncthreads();
  }
  float mu = sm[0] * (1.0f / D); __syncthreads();
  float c = xv - mu;
  sm[d] = c * c; __syncthreads();
  for (int o = D / 2; o > 0; o >>= 1) {
    if (d < o) sm[d] += sm[d + o];
    __syncthreads();
  }
  float var = sm[0] * (1.0f / D);
  float y = c * rsqrtf(var + 1e-5f) * g[d] + bb[d];
  hout[(long)r * D + d] = y;
  h16[(long)r * D + d] = (_Float16)y;
}

// hp[r] = concat(h16[r], (f16)user_emb[active_user[u]])
__global__ void concat_k(const _Float16* __restrict__ h16, const float* __restrict__ user_emb,
                         const int* __restrict__ active_user, _Float16* __restrict__ hp) {
  int r = blockIdx.x, d = threadIdx.x;
  int u = r % U;
  hp[(long)r * (2 * D) + d] = h16[(long)r * D + d];
  hp[(long)r * (2 * D) + D + d] = (_Float16)user_emb[(long)active_user[u] * D + d];
}

// ---------------------------------------------------------------------------
static void launch_gemm(hipStream_t st,
                        const _Float16* A, int adiv, long aso, long asi, int lda,
                        const _Float16* B, int bdiv, long bso, long bsi, int ldb,
                        float* C, _Float16* C16, int cdiv, long cso, long csi, int ldc,
                        const float* bias, int M, int N, int K, float alpha, int relu,
                        int batch) {
  dim3 g((N + 63) / 64, (M + 31) / 32, batch);
  wmma_gemm_f16<<<g, dim3(256), 0, st>>>(A, adiv, aso, asi, lda,
                                         B, bdiv, bso, bsi, ldb,
                                         C, C16, cdiv, cso, csi, ldc,
                                         bias, M, N, K, alpha, relu);
}

extern "C" void kernel_launch(void* const* d_in, const int* in_sizes, int n_in,
                              void* d_out, int out_size, void* d_ws, size_t ws_size,
                              hipStream_t stream) {
  (void)in_sizes; (void)n_in; (void)out_size; (void)ws_size;
  const int*   x          = (const int*)d_in[0];
  const float* t          = (const float*)d_in[1];
  const int*   t_slot     = (const int*)d_in[2];
  const float* scoord     = (const float*)d_in[3];
  const int*   active_usr = (const int*)d_in[7];
  const int*   graph_cols = (const int*)d_in[9];
  const float* graph_vals = (const float*)d_in[10];
  const int*   inter_cols = (const int*)d_in[12];
  const float* inter_vals = (const float*)d_in[13];
  const float* loc_emb    = (const float*)d_in[14];
  const float* user_emb   = (const float*)d_in[15];
  const float* time_emb   = (const float*)d_in[16];
  const float* coord_w    = (const float*)d_in[17];
  const float* coord_b    = (const float*)d_in[18];
  const float* alpha      = (const float*)d_in[19];
  const float* attn_in_w  = (const float*)d_in[20];
  const float* attn_in_b  = (const float*)d_in[21];
  const float* attn_out_w = (const float*)d_in[22];
  const float* attn_out_b = (const float*)d_in[23];
  const float* ln1_g      = (const float*)d_in[24];
  const float* ln1_b      = (const float*)d_in[25];
  const float* ff1_w      = (const float*)d_in[26];
  const float* ff1_b      = (const float*)d_in[27];
  const float* ff2_w      = (const float*)d_in[28];
  const float* ff2_b      = (const float*)d_in[29];
  const float* ln2_g      = (const float*)d_in[30];
  const float* ln2_b      = (const float*)d_in[31];
  const float* fc_w       = (const float*)d_in[32];
  const float* fc_b       = (const float*)d_in[33];
  float* out = (float*)d_out;

  // ---- workspace bump allocator ----
  char* base = (char*)d_ws;
  size_t cur = 0;
  auto alloc = [&](size_t bytes) -> void* {
    void* p = base + cur;
    cur = (cur + bytes + 255) & ~(size_t)255;
    return p;
  };
  const int M = S * U;  // 8192
  _Float16* w_in16  = (_Float16*)alloc((size_t)L * 3 * D * D * 2);
  _Float16* w_out16 = (_Float16*)alloc((size_t)L * D * D * 2);
  _Float16* w_ff116 = (_Float16*)alloc((size_t)L * 4 * D * D * 2);
  _Float16* w_ff216 = (_Float16*)alloc((size_t)L * D * 4 * D * 2);
  _Float16* fcw16   = (_Float16*)alloc((size_t)V * 2 * D * 2);
  _Float16* h16     = (_Float16*)alloc((size_t)M * D * 2);
  _Float16* qkv16   = (_Float16*)alloc((size_t)M * 3 * D * 2);
  _Float16* vT16    = (_Float16*)alloc((size_t)U * H * HD * S * 2);
  _Float16* attn16  = (_Float16*)alloc((size_t)U * H * S * S * 2);
  _Float16* ctx16   = (_Float16*)alloc((size_t)M * D * 2);
  _Float16* ffo16   = (_Float16*)alloc((size_t)M * 4 * D * 2);
  _Float16* hp16    = (_Float16*)alloc((size_t)M * 2 * D * 2);
  float* upref   = (float*)alloc((size_t)U * D * 4);
  float* sim     = (float*)alloc((size_t)U * S * 4);
  float* biasbuf = (float*)alloc((size_t)U * S * S * 4);
  float* h32     = (float*)alloc((size_t)M * D * 4);
  float* sc32    = (float*)alloc((size_t)U * H * S * S * 4);
  float* delta32 = (float*)alloc((size_t)M * D * 4);

  // ---- weight conversion to f16 ----
  auto cvt = [&](const float* src, _Float16* dst, long n) {
    f32_to_f16_k<<<2048, 256, 0, stream>>>(src, dst, n);
  };
  cvt(attn_in_w,  w_in16,  (long)L * 3 * D * D);
  cvt(attn_out_w, w_out16, (long)L * D * D);
  cvt(ff1_w,      w_ff116, (long)L * 4 * D * D);
  cvt(ff2_w,      w_ff216, (long)L * D * 4 * D);
  cvt(fc_w,       fcw16,   (long)V * 2 * D);

  // ---- embedding / bias front-end ----
  user_pref_k<<<U, D, 0, stream>>>(active_usr, inter_cols, inter_vals, loc_emb, upref);
  src_k<<<M, D, 0, stream>>>(x, t_slot, scoord, graph_cols, graph_vals, loc_emb,
                             time_emb, coord_w, coord_b, upref, h32, h16, sim);
  bias_k<<<dim3(S, U), S, 0, stream>>>(t, scoord, sim, alpha, biasbuf);

  const float scale = 0.125f;  // 1/sqrt(64)
  for (int l = 0; l < L; ++l) {
    // qkv = h @ W_in^T + b   (M x 1536)
    launch_gemm(stream, h16, 1, 0, 0, D,
                w_in16 + (long)l * 3 * D * D, 1, 0, 0, D,
                nullptr, qkv16, 1, 0, 0, 3 * D,
                attn_in_b + (long)l * 3 * D, M, 3 * D, D, 1.0f, 0, 1);
    repackV_k<<<M, D, 0, stream>>>(qkv16, vT16);
    // scores[u,h] = Q @ K^T ; batch b = u*H+h, Q/K strided inside qkv16
    launch_gemm(stream, qkv16, H, 3 * D, HD, U * 3 * D,
                qkv16 + D, H, 3 * D, HD, U * 3 * D,
                sc32, nullptr, 1, (long)S * S, 0, S,
                nullptr, S, S, HD, 1.0f, 0, U * H);
    softmax_k<<<dim3(S, U * H), S, 0, stream>>>(sc32, biasbuf, attn16, scale);
    // ctx[u,h] = attn @ V ; via vT (hd x S) as B(NxK)
    launch_gemm(stream, attn16, 1, (long)S * S, 0, S,
                vT16, 1, (long)HD * S, 0, S,
                nullptr, ctx16, H, D, HD, U * D,
                nullptr, S, HD, S, 1.0f, 0, U * H);
    // out proj
    launch_gemm(stream, ctx16, 1, 0, 0, D,
                w_out16 + (long)l * D * D, 1, 0, 0, D,
                delta32, nullptr, 1, 0, 0, D,
                attn_out_b + (long)l * D, M, D, D, 1.0f, 0, 1);
    ln_k<<<M, D, 0, stream>>>(h32, delta32, ln1_g + (long)l * D, ln1_b + (long)l * D, h32, h16);
    // ff1 (relu)
    launch_gemm(stream, h16, 1, 0, 0, D,
                w_ff116 + (long)l * 4 * D * D, 1, 0, 0, D,
                nullptr, ffo16, 1, 0, 0, 4 * D,
                ff1_b + (long)l * 4 * D, M, 4 * D, D, 1.0f, 1, 1);
    // ff2
    launch_gemm(stream, ffo16, 1, 0, 0, 4 * D,
                w_ff216 + (long)l * D * 4 * D, 1, 0, 0, 4 * D,
                delta32, nullptr, 1, 0, 0, D,
                ff2_b + (long)l * D, M, D, 4 * D, 1.0f, 0, 1);
    ln_k<<<M, D, 0, stream>>>(h32, delta32, ln2_g + (long)l * D, ln2_b + (long)l * D, h32, h16);
  }

  // concat [h, p_u] and final FC: (8192 x 1024) @ (1024 x 20000)
  concat_k<<<M, D, 0, stream>>>(h16, user_emb, active_usr, hp16);
  launch_gemm(stream, hp16, 1, 0, 0, 2 * D,
              fcw16, 1, 0, 0, 2 * D,
              out, nullptr, 1, 0, 0, V,
              fc_b, M, V, 2 * D, 1.0f, 0, 1);
}
